// UNetBlockDownsasa_share_69501160783980
// MI455X (gfx1250) — matrix operand order repaired and checked by reading, not compiled, not run on hardware
//
#include <hip/hip_runtime.h>
#include <hip/hip_bf16.h>

typedef __attribute__((ext_vector_type(16))) _Float16 v16h;
typedef __attribute__((ext_vector_type(8)))  _Float16 v8h;
typedef __attribute__((ext_vector_type(8)))  float    v8f;
typedef __attribute__((ext_vector_type(4)))  unsigned int u32x4;
typedef __attribute__((ext_vector_type(8)))  int      i32x8;
typedef __attribute__((ext_vector_type(4)))  int      i32x4;

#define HEADS 4
#define COUT  128
#define BNEPS 1e-5f

#if __has_builtin(__builtin_amdgcn_tensor_load_to_lds) && \
    __has_builtin(__builtin_amdgcn_s_wait_tensorcnt)
#define TDM_OK 1
#endif

static __device__ __forceinline__ float lrelu_f(float x) { return x > 0.f ? x : 0.05f * x; }

// ---------------------------------------------------------------------------
// Weight / bias preparation (folds residual paths into the 3x3 center tap).
// ---------------------------------------------------------------------------
__global__ void make_weights_kernel(
    const float* __restrict__ Wadj, const float* __restrict__ badj,
    const float* __restrict__ Wc0,  const float* __restrict__ bconv0,
    const float* __restrict__ Wc1,  const float* __restrict__ Wq,
    const float* __restrict__ Wkv,  const float* __restrict__ Wp,
    const float* __restrict__ g1,   const float* __restrict__ be1,
    const float* __restrict__ m1,   const float* __restrict__ v1,
    _Float16* __restrict__ wc0o, _Float16* __restrict__ wc1o,
    _Float16* __restrict__ wqo,  _Float16* __restrict__ wkvo,
    _Float16* __restrict__ wpo,
    float* __restrict__ bc0o, float* __restrict__ s1o, float* __restrict__ t1o)
{
  const int idx = blockIdx.x * blockDim.x + threadIdx.x;
  const int K0 = 64 * 9, K1 = 128 * 9;

  if (idx < 128 * K1) {
    int o = idx / K1, k = idx % K1;
    int kid = k / 128, ci = k % 128;
    int kh = kid / 3, kw = kid % 3;
    float w = Wc1[((o * 128 + ci) * 3 + kh) * 3 + kw];
    if (kid == 4 && o == ci) w += 1.0f;                 // identity residual
    wc1o[o * K1 + k] = (_Float16)w;
    wpo[o * K1 + k]  = (_Float16)Wp[((o * 128 + ci) * 3 + kh) * 3 + kw];
  }
  if (idx < 128 * K0) {
    int o = idx / K0, k = idx % K0;
    int kid = k / 64, ci = k % 64;
    int kh = kid / 3, kw = kid % 3;
    float w = Wc0[((o * 64 + ci) * 3 + kh) * 3 + kw];
    if (kid == 4) w += Wadj[o * 64 + ci];               // 1x1 adjust residual
    wc0o[o * K0 + k] = (_Float16)w;
  }
  if (idx < 128 * 128) {
    wqo[idx]  = (_Float16)Wq[idx];
    wkvo[idx] = (_Float16)Wkv[idx];
  }
  if (idx < 128) {
    bc0o[idx] = bconv0[idx] + badj[idx];
    float s = g1[idx] * rsqrtf(v1[idx] + BNEPS);
    s1o[idx] = s;
    t1o[idx] = be1[idx] - m1[idx] * s;
  }
}

// ---------------------------------------------------------------------------
__global__ void bn_cast_kernel(const float* __restrict__ x,
                               const float* __restrict__ g,  const float* __restrict__ be,
                               const float* __restrict__ mu, const float* __restrict__ var,
                               _Float16* __restrict__ out, int C, int HW, int total)
{
  int i = blockIdx.x * blockDim.x + threadIdx.x;
  if (i >= total) return;
  int c = (i / HW) % C;
  float s = g[c] * rsqrtf(var[c] + BNEPS);
  out[i] = (_Float16)((x[i] - mu[c]) * s + be[c]);
}

// ---------------------------------------------------------------------------
__global__ void kv_border_kernel(_Float16* __restrict__ kvp,
                                 const float* __restrict__ bkv, int total)
{
  int i = blockIdx.x * blockDim.x + threadIdx.x;
  if (i >= total) return;
  const int P = 130;
  int w = i % P;
  int t = i / P;
  int h = t % P;
  int c = (t / P) % 128;
  if (h == 0 || h == P - 1 || w == 0 || w == P - 1) kvp[i] = (_Float16)bkv[c];
}

// ---------------------------------------------------------------------------
// Implicit-GEMM conv via v_wmma_f32_16x16x32_f16.
//   M=128 channels, N=B*outW*outW pixels (outW power of two, oShift=log2),
//   K=ks*ks*Cin. 256 threads = 8 wave32; block tile 128(M) x 64(N);
//   wave tile 32x32 = 2x2 WMMA accumulators.
// A (weights) staged to LDS by the Tensor Data Mover when available;
// B staged transposed [n][k] so fragments are contiguous b128 LDS loads.
// ---------------------------------------------------------------------------
__global__ __launch_bounds__(256)
void conv_wmma_kernel(const _Float16* __restrict__ act,
                      const _Float16* __restrict__ wgt,
                      const float* __restrict__ bias,
                      const float* __restrict__ bnS,
                      const float* __restrict__ bnT,
                      float* __restrict__ outF,
                      _Float16* __restrict__ out16,
                      int B, int Cin, int H, int W,
                      int ks, int pad, int stride,
                      int oShift, int outPH, int outPW,
                      int rOff, int cOff, int doLrelu)
{
  __shared__ __align__(16) _Float16 ldsA[128 * 32];   // weights   M x Kc (row major)
  __shared__ __align__(16) _Float16 ldsBT[64 * 32];   // im2col    N x Kc (transposed)

  const int K     = ks * ks * Cin;
  const int HW    = H * W;
  const int outW  = 1 << oShift;
  const int nBase = blockIdx.x * 64;
  const int lane  = threadIdx.x & 31;
  const int wave  = threadIdx.x >> 5;
  const int half  = lane >> 4;
  const int lan   = lane & 15;
  const int wM    = (wave & 3) * 32;
  const int wN    = (wave >> 2) * 32;

  v8f acc[2][2];
#pragma unroll
  for (int mi = 0; mi < 2; ++mi)
#pragma unroll
    for (int ni = 0; ni < 2; ++ni)
#pragma unroll
      for (int e = 0; e < 8; ++e) acc[mi][ni][e] = 0.f;

  // ---- per-thread im2col coordinates (K-invariant), 8 pixels each ----
  const int bkk = threadIdx.x & 31;   // K index within chunk
  const int bng = threadIdx.x >> 5;   // pixel group
  int ih0[8], iw0[8], bof[8];
#pragma unroll
  for (int j = 0; j < 8; ++j) {
    const int n  = nBase + bng * 8 + j;
    const int ow = n & (outW - 1);
    const int oh = (n >> oShift) & (outW - 1);
    const int bb = n >> (2 * oShift);
    ih0[j] = oh * stride;
    iw0[j] = ow * stride;
    bof[j] = bb * Cin * HW;
  }
#if !defined(TDM_OK)
  const int arow = threadIdx.x >> 1;
  const int aseg = (threadIdx.x & 1) * 16;
#endif

  for (int kid = 0; kid < ks * ks; ++kid) {
    const int dh = kid / ks - pad;
    const int dw = kid % ks - pad;
    for (int cc = 0; cc < Cin; cc += 32) {
      // ---- stage A: 128x32 f16 weight tile ----
#if defined(TDM_OK)
      if (threadIdx.x < 32) {
        // Tensor DMA descriptor: 2D tile, 32 elements/row (2B), 128 rows,
        // row stride = K elements, packed row-major into ldsA.
        const unsigned long long ga =
            (unsigned long long)(size_t)(wgt + (size_t)kid * Cin + cc);
        const unsigned int ldsOff = (unsigned int)(size_t)(void*)ldsA;
        u32x4 g0;
        g0[0] = 1u;                                        // count=1, user mode
        g0[1] = ldsOff;                                    // lds_addr
        g0[2] = (unsigned int)(ga & 0xffffffffu);          // global_addr[31:0]
        g0[3] = (unsigned int)((ga >> 32) & 0x01ffffffu)   // global_addr[56:32]
                | (2u << 30);                              // type=2 (image)
        i32x8 g1;
        g1[0] = (int)(1u << 16);                           // data_size=1 (2 bytes)
        g1[1] = (int)(32u << 16);                          // tensor_dim0 = 32
        g1[2] = (int)(128u << 16);                         // tensor_dim1 = 128
        g1[3] = (int)(32u << 16);                          // tile_dim0 = 32
        g1[4] = 128;                                       // tile_dim1 = 128
        g1[5] = K;                                         // tensor_dim0_stride
        g1[6] = 0;
        g1[7] = 0;
        i32x4 gz;
        gz[0] = 0; gz[1] = 0; gz[2] = 0; gz[3] = 0;
#if __clang_major__ >= 23
        i32x8 gz8;
#pragma unroll
        for (int e = 0; e < 8; ++e) gz8[e] = 0;
        __builtin_amdgcn_tensor_load_to_lds(g0, g1, gz, gz, gz8, 0);
#else
        __builtin_amdgcn_tensor_load_to_lds(g0, g1, gz, gz, 0);
#endif
      }
#else
      {
        const uint4* src = (const uint4*)(wgt + (size_t)arow * K + kid * Cin + cc + aseg);
        uint4* dst = (uint4*)(ldsA + arow * 32 + aseg);
        dst[0] = src[0];
        dst[1] = src[1];
      }
#endif
      // ---- stage B: 64(N) x 32(K) f16 im2col (transposed), zero padded ----
      {
        const int ci = cc + bkk;
#pragma unroll
        for (int j = 0; j < 8; ++j) {
          const int ih = ih0[j] + dh;
          const int iw = iw0[j] + dw;
          _Float16 v = (_Float16)0.f;
          if ((unsigned)ih < (unsigned)H && (unsigned)iw < (unsigned)W)
            v = act[(size_t)bof[j] + (size_t)ci * HW + ih * W + iw];
          ldsBT[(bng * 8 + j) * 32 + bkk] = v;
        }
      }
#if defined(TDM_OK)
      if (threadIdx.x < 32) __builtin_amdgcn_s_wait_tensorcnt(0);
#endif
      __syncthreads();

      // ---- fragments: contiguous b128 LDS loads per documented layouts ----
      v16h Af[2], Bf[2];
#pragma unroll
      for (int mi = 0; mi < 2; ++mi) {
        const _Float16* row = ldsA + (wM + mi * 16 + lan) * 32;
        const v8h lo = *(const v8h*)(row + half * 8);        // k = half*8 .. +7
        const v8h hi = *(const v8h*)(row + 16 + half * 8);   // k = 16+half*8 .. +7
        Af[mi] = __builtin_shufflevector(lo, hi, 0, 1, 2, 3, 4, 5, 6, 7,
                                         8, 9, 10, 11, 12, 13, 14, 15);
      }
#pragma unroll
      for (int ni = 0; ni < 2; ++ni) {
        const _Float16* row = ldsBT + (wN + ni * 16 + lan) * 32;
        const v8h lo = *(const v8h*)(row + half * 8);        // k = half*8 .. +7
        const v8h hi = *(const v8h*)(row + 16 + half * 8);   // k = 16+half*8 .. +7
        Bf[ni] = __builtin_shufflevector(lo, hi, 0, 1, 2, 3, 4, 5, 6, 7,
                                         8, 9, 10, 11, 12, 13, 14, 15);
      }

#pragma unroll
      for (int mi = 0; mi < 2; ++mi)
#pragma unroll
        for (int ni = 0; ni < 2; ++ni)
          acc[mi][ni] = __builtin_amdgcn_wmma_f32_16x16x32_f16(
              false, Af[mi], false, Bf[ni], (short)0, acc[mi][ni], false, false);

      __syncthreads();
    }
  }

  // ---- epilogue: C layout row m = half*8 + v, col n = lane&15 ----
#pragma unroll
  for (int mi = 0; mi < 2; ++mi) {
#pragma unroll
    for (int ni = 0; ni < 2; ++ni) {
      const int n  = nBase + wN + ni * 16 + lan;
      const int ow = n & (outW - 1);
      const int oh = (n >> oShift) & (outW - 1);
      const int bb = n >> (2 * oShift);
#pragma unroll
      for (int v = 0; v < 8; ++v) {
        const int co = wM + mi * 16 + half * 8 + v;
        float val = acc[mi][ni][v] + bias[co];
        if (doLrelu) val = lrelu_f(val);
        const size_t oidx =
            (((size_t)bb * COUT + co) * outPH + (oh + rOff)) * outPW + (ow + cOff);
        if (outF) outF[oidx] = val;
        if (out16) {
          float g = val;
          if (bnS) g = g * bnS[co] + bnT[co];   // folded BN1 before f16 store
          out16[oidx] = (_Float16)g;
        }
      }
    }
  }
}

// ---------------------------------------------------------------------------
// SASA: per (batch, head, pixel) 9-neighbor attention; out = mean of branches.
// ---------------------------------------------------------------------------
__global__ __launch_bounds__(256)
void sasa_kernel(const _Float16* __restrict__ q16,
                 const _Float16* __restrict__ kva,
                 const _Float16* __restrict__ kvb,
                 float* __restrict__ outF,
                 _Float16* __restrict__ out16,
                 int B, int H, int W)
{
  const int HW = H * W;
  const int idx = blockIdx.x * blockDim.x + threadIdx.x;
  if (idx >= B * HEADS * HW) return;
  const int pix = idx % HW;
  const int hd  = (idx / HW) % HEADS;
  const int b   = idx / (HW * HEADS);
  const int h = pix / W, w = pix % W;
  const int P = W + 2;
  const float scale = 0.17677669529663687f;   // (d=32)^-0.5

  float q[32];
#pragma unroll
  for (int dd = 0; dd < 32; ++dd)
    q[dd] = (float)q16[(((size_t)b * COUT + hd * 32 + dd) * H + h) * W + w];

  float acc[32];
#pragma unroll
  for (int dd = 0; dd < 32; ++dd) acc[dd] = 0.f;

  for (int br = 0; br < 2; ++br) {
    const _Float16* kv = br ? kvb : kva;
    const size_t base = ((size_t)b * COUT + hd * 32) * (size_t)(H + 2) * (size_t)P
                        + (size_t)h * P + w;
    float lg[9];
#pragma unroll
    for (int y = 0; y < 9; ++y) {
      const int di = y / 3, dj = y % 3;
      float s = 0.f;
#pragma unroll
      for (int dd = 0; dd < 32; ++dd)
        s += q[dd] * (float)kv[base + (size_t)dd * (H + 2) * P + di * P + dj];
      lg[y] = s * scale;
    }
    float mx = lg[0];
#pragma unroll
    for (int y = 1; y < 9; ++y) mx = fmaxf(mx, lg[y]);
    float sum = 0.f;
#pragma unroll
    for (int y = 0; y < 9; ++y) { lg[y] = __expf(lg[y] - mx); sum += lg[y]; }
    const float inv = 1.f / sum;
#pragma unroll
    for (int y = 0; y < 9; ++y) {
      const int di = y / 3, dj = y % 3;
      const float wg = lg[y] * inv;
#pragma unroll
      for (int dd = 0; dd < 32; ++dd)
        acc[dd] += wg * (float)kv[base + (size_t)dd * (H + 2) * P + di * P + dj];
    }
  }
#pragma unroll
  for (int dd = 0; dd < 32; ++dd) {
    const float val = acc[dd] * 0.5f;
    const size_t o = (((size_t)b * COUT + hd * 32 + dd) * H + h) * W + w;
    outF[o]  = val;
    out16[o] = (_Float16)val;
  }
}

// ---------------------------------------------------------------------------
extern "C" void kernel_launch(void* const* d_in, const int* in_sizes, int n_in,
                              void* d_out, int out_size, void* d_ws, size_t ws_size,
                              hipStream_t stream)
{
  (void)in_sizes; (void)n_in; (void)out_size; (void)ws_size;
  const int B = 2, CIN = 64, H = 128, W = 128, HW = H * W;
  const int K0 = 64 * 9, K1 = 128 * 9;

  const float* x[3]   = {(const float*)d_in[0], (const float*)d_in[1], (const float*)d_in[2]};
  const float* g0     = (const float*)d_in[3];
  const float* be0    = (const float*)d_in[4];
  const float* mu0    = (const float*)d_in[5];
  const float* va0    = (const float*)d_in[6];
  const float* g1     = (const float*)d_in[7];
  const float* be1    = (const float*)d_in[8];
  const float* mu1    = (const float*)d_in[9];
  const float* va1    = (const float*)d_in[10];
  const float* Wadj   = (const float*)d_in[11];
  const float* badj   = (const float*)d_in[12];
  const float* Wconv0 = (const float*)d_in[13];
  const float* bconv0 = (const float*)d_in[14];
  const float* Wconv1 = (const float*)d_in[15];
  const float* bconv1 = (const float*)d_in[16];
  const float* Wqm    = (const float*)d_in[17];
  const float* bq     = (const float*)d_in[18];
  const float* Wkvm   = (const float*)d_in[19];
  const float* bkv    = (const float*)d_in[20];
  const float* Wpool  = (const float*)d_in[21];
  const float* bpool  = (const float*)d_in[22];

  char* ws = (char*)d_ws;
  size_t off = 0;
  auto alloc = [&](size_t bytes) -> char* {
    char* p = ws + off;
    off = (off + bytes + 255) & ~(size_t)255;
    return p;
  };
  _Float16* wc0p = (_Float16*)alloc((size_t)128 * K0 * 2);
  _Float16* wc1p = (_Float16*)alloc((size_t)128 * K1 * 2);
  _Float16* wqp  = (_Float16*)alloc((size_t)128 * 128 * 2);
  _Float16* wkvp = (_Float16*)alloc((size_t)128 * 128 * 2);
  _Float16* wpp  = (_Float16*)alloc((size_t)128 * K1 * 2);
  float*    bc0p = (float*)alloc(128 * 4);
  float*    s1p  = (float*)alloc(128 * 4);
  float*    t1p  = (float*)alloc(128 * 4);
  _Float16* xb0p = (_Float16*)alloc((size_t)B * CIN * HW * 2);
  _Float16* a16p = (_Float16*)alloc((size_t)B * COUT * HW * 2);
  _Float16* n16p[3];
  for (int t = 0; t < 3; ++t) n16p[t] = (_Float16*)alloc((size_t)B * COUT * HW * 2);
  _Float16* q16p = (_Float16*)alloc((size_t)B * COUT * HW * 2);
  _Float16* kv1p = (_Float16*)alloc((size_t)B * COUT * 130 * 130 * 2);
  _Float16* kv2p = (_Float16*)alloc((size_t)B * COUT * 130 * 130 * 2);
  _Float16* s16p = (_Float16*)alloc((size_t)B * COUT * HW * 2);

  float* outp      = (float*)d_out;
  const int poolSz = B * COUT * 64 * 64;
  float* out_n0    = outp;
  float* out_n1    = outp + poolSz;
  float* out_n2    = outp + 2 * poolSz;
  float* out_pass  = outp + 3 * poolSz;

  // 1) weights / folded constants
  make_weights_kernel<<<(128 * K1 + 255) / 256, 256, 0, stream>>>(
      Wadj, badj, Wconv0, bconv0, Wconv1, Wqm, Wkvm, Wpool,
      g1, be1, mu1, va1, wc0p, wc1p, wqp, wkvp, wpp, bc0p, s1p, t1p);

  // 2) rep chain per tensor
  const int convTiles = (B * HW) / 64;          // 512
  for (int t = 0; t < 3; ++t) {
    const int tot0 = B * CIN * HW;
    bn_cast_kernel<<<(tot0 + 255) / 256, 256, 0, stream>>>(
        x[t], g0, be0, mu0, va0, xb0p, CIN, HW, tot0);
    conv_wmma_kernel<<<convTiles, 256, 0, stream>>>(
        xb0p, wc0p, bc0p, s1p, t1p, nullptr, a16p,
        B, CIN, H, W, 3, 1, 1, /*oShift*/7, H, W, 0, 0, 1);
    conv_wmma_kernel<<<convTiles, 256, 0, stream>>>(
        a16p, wc1p, bconv1, nullptr, nullptr, nullptr, n16p[t],
        B, COUT, H, W, 3, 1, 1, 7, H, W, 0, 0, 1);
  }

  // 3) SASA projections: q from n1; kv from n0 / n2 into padded buffers
  conv_wmma_kernel<<<convTiles, 256, 0, stream>>>(
      n16p[1], wqp, bq, nullptr, nullptr, nullptr, q16p,
      B, COUT, H, W, 1, 0, 1, 7, H, W, 0, 0, 0);
  const int kvTot = B * COUT * 130 * 130;
  kv_border_kernel<<<(kvTot + 255) / 256, 256, 0, stream>>>(kv1p, bkv, kvTot);
  kv_border_kernel<<<(kvTot + 255) / 256, 256, 0, stream>>>(kv2p, bkv, kvTot);
  conv_wmma_kernel<<<convTiles, 256, 0, stream>>>(
      n16p[0], wkvp, bkv, nullptr, nullptr, nullptr, kv1p,
      B, COUT, H, W, 1, 0, 1, 7, 130, 130, 1, 1, 0);
  conv_wmma_kernel<<<convTiles, 256, 0, stream>>>(
      n16p[2], wkvp, bkv, nullptr, nullptr, nullptr, kv2p,
      B, COUT, H, W, 1, 0, 1, 7, 130, 130, 1, 1, 0);

  // 4) attention -> pass_layer (f32) + f16 copy for pooling
  const int attTot = B * HEADS * HW;
  sasa_kernel<<<(attTot + 255) / 256, 256, 0, stream>>>(
      q16p, kv1p, kv2p, out_pass, s16p, B, H, W);

  // 5) pooling convs (3x3, stride 2, pad 1)
  const int poolTiles = (B * 64 * 64) / 64;     // 128
  conv_wmma_kernel<<<poolTiles, 256, 0, stream>>>(
      n16p[0], wpp, bpool, nullptr, nullptr, out_n0, nullptr,
      B, COUT, H, W, 3, 1, 2, /*oShift*/6, 64, 64, 0, 0, 0);
  conv_wmma_kernel<<<poolTiles, 256, 0, stream>>>(
      s16p, wpp, bpool, nullptr, nullptr, out_n1, nullptr,
      B, COUT, H, W, 3, 1, 2, 6, 64, 64, 0, 0, 0);
  conv_wmma_kernel<<<poolTiles, 256, 0, stream>>>(
      n16p[2], wpp, bpool, nullptr, nullptr, out_n2, nullptr,
      B, COUT, H, W, 3, 1, 2, 6, 64, 64, 0, 0, 0);
}
